// NeuralODE_87162066305758
// MI455X (gfx1250) — matrix operand order repaired and checked by reading, not compiled, not run on hardware
//
#include <hip/hip_runtime.h>

// ---------------------------------------------------------------------------
// Neural ODE (RK45 Dormand-Prince) persistent-workgroup kernel for gfx1250.
// B=1024, D=256, H=512, T=11, SUBSTEPS=4  =>  h = 10/(10*4) = 0.25
// Each workgroup integrates BM=16 batch rows for the full trajectory.
// GEMMs via v_wmma_f32_16x16x32_bf16; weights pre-converted to bf16 in d_ws.
// Software-pipelined k-loop: next k-step's A/B fragments are prefetched into
// a second register buffer while the current WMMAs execute, so s_wait_loadcnt
// before each WMMA group covers a full k-step of latency.
// ---------------------------------------------------------------------------

typedef __attribute__((ext_vector_type(16))) __bf16   v16bf;
typedef __attribute__((ext_vector_type(8)))  float    v8f;
typedef __attribute__((ext_vector_type(4)))  unsigned u32x4;
typedef __attribute__((ext_vector_type(8)))  unsigned u32x8;

// Address-space qualified element types (1 = global, 3 = LDS).
typedef unsigned short AS1u16  __attribute__((address_space(1)));
typedef unsigned short AS3u16  __attribute__((address_space(3)));
typedef float          AS3f32  __attribute__((address_space(3)));
typedef u32x4          AS1u32x4 __attribute__((address_space(1)));
typedef u32x4          AS3u32x4 __attribute__((address_space(3)));

#define DD      256
#define HH      512
#define BM      16
#define THREADS 256
#define TOUT    10
#define HS      0.25f
#define YSTR    (DD + 8)   // padded LDS row strides (break bank conflicts)
#define HSTR    (HH + 8)

__device__ __forceinline__ unsigned short f2bf(float f) {
    unsigned u = __builtin_bit_cast(unsigned, f);
    u += 0x7FFFu + ((u >> 16) & 1u);            // round-to-nearest-even
    return (unsigned short)(u >> 16);
}

// out[m, n] = act(in[m, :] . W[n, :] + bias[n])    (one 16-row M tile)
// k-outer / nt-inner with one-stage software pipeline:
//   - A fragment (LDS) loaded once per k-step, shared by all NTILES tiles
//   - B fragments (global, L2-resident) prefetched one k-step ahead
template<int K, int INSTR, int N, int OUTSTR, int NTILES, bool RELU, bool OUTF32>
__device__ __forceinline__ void gemm_layer(
    const AS3u16* __restrict__ inAct,   // LDS  BM x K (stride INSTR)
    const AS1u16* __restrict__ Wg,      // glob N  x K bf16, row-major
    const AS3f32* __restrict__ bias,    // LDS  N
    AS3u16*       __restrict__ outBF,   // LDS  BM x N (stride OUTSTR)
    AS3f32*       __restrict__ outF,    // LDS  BM x N (stride OUTSTR)
    int wave, int lane)
{
    const int r = lane & 15;      // A: row / B,C,D: column-within-tile
    const int c = lane >> 4;      // K-half select / C row-half select

    const AS3u16* aBase = inAct + r * INSTR + c * 8;
    const AS1u16* bBase = Wg + (size_t)(wave * (NTILES * 16) + r) * K + c * 8;

    v8f acc[NTILES];
#pragma unroll
    for (int nt = 0; nt < NTILES; ++nt)
        acc[nt] = (v8f){0.f, 0.f, 0.f, 0.f, 0.f, 0.f, 0.f, 0.f};

    // Prologue: fragments for k0 = 0.
    u32x8 aCur, bCur[NTILES];
    aCur.lo = *(const AS3u32x4*)(aBase);
    aCur.hi = *(const AS3u32x4*)(aBase + 16);
#pragma unroll
    for (int nt = 0; nt < NTILES; ++nt) {
        const AS1u16* bp = bBase + (size_t)nt * 16 * K;
        bCur[nt].lo = *(const AS1u32x4*)(bp);
        bCur[nt].hi = *(const AS1u32x4*)(bp + 16);
    }

#pragma unroll
    for (int k0 = 0; k0 < K; k0 += 32) {
        const bool more = (k0 + 32) < K;          // compile-time per iteration
        u32x8 aNxt, bNxt[NTILES];
        if (more) {
            const AS3u16* ap = aBase + (k0 + 32);
            aNxt.lo = *(const AS3u32x4*)(ap);
            aNxt.hi = *(const AS3u32x4*)(ap + 16);
#pragma unroll
            for (int nt = 0; nt < NTILES; ++nt) {
                const AS1u16* bp = bBase + (size_t)nt * 16 * K + (k0 + 32);
                bNxt[nt].lo = *(const AS1u32x4*)(bp);
                bNxt[nt].hi = *(const AS1u32x4*)(bp + 16);
            }
        }
        const v16bf av = __builtin_bit_cast(v16bf, aCur);
#pragma unroll
        for (int nt = 0; nt < NTILES; ++nt)
            acc[nt] = __builtin_amdgcn_wmma_f32_16x16x32_bf16(
                false, av, false, __builtin_bit_cast(v16bf, bCur[nt]),
                (short)0, acc[nt], false, false);
        if (more) {
            aCur = aNxt;
#pragma unroll
            for (int nt = 0; nt < NTILES; ++nt) bCur[nt] = bNxt[nt];
        }
    }

    // C/D f32 layout: lane<16 -> N=lane, M=vgpr; lane>=16 -> M=8+vgpr
    const int mb = c * 8;
#pragma unroll
    for (int nt = 0; nt < NTILES; ++nt) {
        const int n = wave * (NTILES * 16) + nt * 16 + r;
        const float bv = bias[n];
#pragma unroll
        for (int i = 0; i < 8; ++i) {
            float v = acc[nt][i] + bv;
            if (RELU) v = v > 0.f ? v : 0.f;
            if (OUTF32) outF [(mb + i) * OUTSTR + n] = v;
            else        outBF[(mb + i) * OUTSTR + n] = f2bf(v);
        }
    }
}

// One MLP evaluation: kout = W3.relu(W2.relu(W1.y+b1)+b2)+b3 for a 16-row tile.
// noinline keeps the 6-stage RK body compact (128 WMMAs live here once).
__device__ __attribute__((noinline)) void mlp_eval(
    const AS3u16* __restrict__ ybf,
    AS3u16* __restrict__ h1, AS3u16* __restrict__ h2,
    AS3f32* __restrict__ kout,
    const AS1u16* __restrict__ W1, const AS1u16* __restrict__ W2,
    const AS1u16* __restrict__ W3, const AS3f32* __restrict__ biasL,
    int wave, int lane)
{
    gemm_layer<DD, YSTR, HH, HSTR, 4, true,  false>(ybf, W1, biasL,        h1, nullptr, wave, lane);
    __syncthreads();
    gemm_layer<HH, HSTR, HH, HSTR, 4, true,  false>(h1,  W2, biasL + HH,   h2, nullptr, wave, lane);
    __syncthreads();
    gemm_layer<HH, HSTR, DD, DD,   2, false, true >(h2,  W3, biasL + 2*HH, nullptr, kout, wave, lane);
    __syncthreads();
}

#define LOADK(dst)                                   \
    { _Pragma("unroll")                              \
      for (int j = 0; j < BM; ++j) dst[j] = kbuf[j * DD + tid]; }

#define STORE_YBF(expr)                              \
    { _Pragma("unroll")                              \
      for (int j = 0; j < BM; ++j) {                 \
          float nv_ = (expr);                        \
          ybf[j * YSTR + tid] = f2bf(nv_);           \
      }                                              \
      __syncthreads(); }

__global__ __launch_bounds__(THREADS) void node_kernel(
    const float* __restrict__ x0,
    const unsigned short* __restrict__ wbf,   // bf16 W1|W2|W3 in d_ws
    const float* __restrict__ b1, const float* __restrict__ b2,
    const float* __restrict__ b3,
    float* __restrict__ out)                  // (B, TOUT, D)
{
    __shared__ unsigned short ybf[BM * YSTR];   //  8.25 KB  stage input (bf16)
    __shared__ unsigned short h1 [BM * HSTR];   // 16.25 KB
    __shared__ unsigned short h2 [BM * HSTR];   // 16.25 KB
    __shared__ float          kbuf[BM * DD];    // 16    KB  f(y) scratch
    __shared__ float          biasL[HH + HH + DD]; // 5 KB  b1|b2|b3

    const int tid  = threadIdx.x;
    const int lane = tid & 31;
    const int wave = tid >> 5;

    // Stage biases into LDS once (reused by all 240 MLP evals).
    for (int i = tid; i < HH + HH + DD; i += THREADS) {
        float v;
        if (i < HH)            v = b1[i];
        else if (i < 2 * HH)   v = b2[i - HH];
        else                   v = b3[i - 2 * HH];
        biasL[i] = v;
    }

    // Address-space-qualified views.
    const AS1u16* W1 = (const AS1u16*)wbf;
    const AS1u16* W2 = W1 + HH * DD;
    const AS1u16* W3 = W1 + HH * DD + HH * HH;
    AS3u16* ybfL = (AS3u16*)ybf;
    AS3u16* h1L  = (AS3u16*)h1;
    AS3u16* h2L  = (AS3u16*)h2;
    AS3f32* kbL  = (AS3f32*)kbuf;
    const AS3f32* bL = (const AS3f32*)biasL;

    // Thread tid owns column d=tid for all 16 rows (i = j*256 + tid).
    float yreg[BM];
    float k1p[BM], k2p[BM], k3p[BM], k4p[BM], k5p[BM], k6p[BM];

    const size_t base = (size_t)blockIdx.x * (BM * DD);
#pragma unroll
    for (int j = 0; j < BM; ++j) {
        float v = x0[base + (size_t)j * DD + tid];
        yreg[j] = v;
        ybf[j * YSTR + tid] = f2bf(v);
    }
    __syncthreads();

    for (int t = 0; t < TOUT; ++t) {
        for (int s = 0; s < 4; ++s) {
            mlp_eval(ybfL, h1L, h2L, kbL, W1, W2, W3, bL, wave, lane);
            LOADK(k1p);
            STORE_YBF(yreg[j] + HS * (0.2f * k1p[j]));

            mlp_eval(ybfL, h1L, h2L, kbL, W1, W2, W3, bL, wave, lane);
            LOADK(k2p);
            STORE_YBF(yreg[j] + HS * ((3.f/40.f)*k1p[j] + (9.f/40.f)*k2p[j]));

            mlp_eval(ybfL, h1L, h2L, kbL, W1, W2, W3, bL, wave, lane);
            LOADK(k3p);
            STORE_YBF(yreg[j] + HS * ((44.f/45.f)*k1p[j] - (56.f/15.f)*k2p[j]
                                    + (32.f/9.f)*k3p[j]));

            mlp_eval(ybfL, h1L, h2L, kbL, W1, W2, W3, bL, wave, lane);
            LOADK(k4p);
            STORE_YBF(yreg[j] + HS * ((19372.f/6561.f)*k1p[j] - (25360.f/2187.f)*k2p[j]
                                    + (64448.f/6561.f)*k3p[j] - (212.f/729.f)*k4p[j]));

            mlp_eval(ybfL, h1L, h2L, kbL, W1, W2, W3, bL, wave, lane);
            LOADK(k5p);
            STORE_YBF(yreg[j] + HS * ((9017.f/3168.f)*k1p[j] - (355.f/33.f)*k2p[j]
                                    + (46732.f/5247.f)*k3p[j] + (49.f/176.f)*k4p[j]
                                    - (5103.f/18656.f)*k5p[j]));

            mlp_eval(ybfL, h1L, h2L, kbL, W1, W2, W3, bL, wave, lane);
            LOADK(k6p);
#pragma unroll
            for (int j = 0; j < BM; ++j) {
                yreg[j] += HS * ((35.f/384.f)*k1p[j] + (500.f/1113.f)*k3p[j]
                               + (125.f/192.f)*k4p[j] - (2187.f/6784.f)*k5p[j]
                               + (11.f/84.f)*k6p[j]);
                ybf[j * YSTR + tid] = f2bf(yreg[j]);
            }
            __syncthreads();
        }
        // Emit y after this interval: out[b, t, d]
#pragma unroll
        for (int j = 0; j < BM; ++j) {
            size_t b = (size_t)blockIdx.x * BM + j;
            out[(b * TOUT + t) * DD + tid] = yreg[j];
        }
    }
}

// One-shot fp32 -> bf16 weight conversion into d_ws.
__global__ __launch_bounds__(THREADS) void convert_weights(
    const float* __restrict__ W1, const float* __restrict__ W2,
    const float* __restrict__ W3, unsigned short* __restrict__ outw)
{
    const int N1 = HH * DD;        // 131072
    const int N2 = HH * HH;        // 262144
    int i = blockIdx.x * THREADS + threadIdx.x;   // total 524288
    float v;
    if (i < N1)            v = W1[i];
    else if (i < N1 + N2)  v = W2[i - N1];
    else                   v = W3[i - N1 - N2];
    outw[i] = f2bf(v);
}

extern "C" void kernel_launch(void* const* d_in, const int* in_sizes, int n_in,
                              void* d_out, int out_size, void* d_ws, size_t ws_size,
                              hipStream_t stream) {
    (void)in_sizes; (void)n_in; (void)out_size; (void)ws_size;
    const float* x0 = (const float*)d_in[0];
    // d_in[1] is T (== 11, baked in as compile-time constant)
    const float* W1 = (const float*)d_in[2];
    const float* b1 = (const float*)d_in[3];
    const float* W2 = (const float*)d_in[4];
    const float* b2 = (const float*)d_in[5];
    const float* W3 = (const float*)d_in[6];
    const float* b3 = (const float*)d_in[7];
    unsigned short* wbf = (unsigned short*)d_ws;   // ~1 MB bf16 weights

    convert_weights<<<(HH*DD + HH*HH + DD*HH) / THREADS, THREADS, 0, stream>>>(
        W1, W2, W3, wbf);
    node_kernel<<<1024 / BM, THREADS, 0, stream>>>(
        x0, wbf, b1, b2, b3, (float*)d_out);
}
